// EnhancedFlowLayer_22101901705433
// MI455X (gfx1250) — matrix operand order repaired and checked by reading, not compiled, not run on hardware
//
#include <hip/hip_runtime.h>
#include <math.h>

// ---------------------------------------------------------------------------
// EnhancedFlowLayer for MI455X (gfx1250, wave32, WMMA, 320KB LDS/WGP)
// B=2, S=512, D=256, H=8, dh=32, P=16, MEM=512
// ---------------------------------------------------------------------------

#define BB   2
#define SS   512
#define DD   256
#define HH   8
#define DH   32
#define PP   16
#define MEMN 512
#define NTOK (BB*SS)

typedef _Float16 f16;
typedef __attribute__((ext_vector_type(16))) _Float16 v16h;
typedef __attribute__((ext_vector_type(8)))  _Float16 v8h;
typedef __attribute__((ext_vector_type(8)))  float    v8f;

// A-operand (16x32 f16) per-lane K index for slot pair r (0..7), lane half.
// Slot s (0..15) of the v16h maps to K = half*16 + s.
__device__ __forceinline__ int aidx(int r, int half) {
    return (half << 4) + (r << 1);
}
// B-operand (32x16 f16): slot s maps to K = half*16 + s  (16 contiguous halves).
__device__ __forceinline__ int bidx(int r, int half) {
    return (half << 4) + (r << 1);
}

__device__ __forceinline__ float gelu_exact(float x) {
    return 0.5f * x * (1.0f + erff(x * 0.70710678118654752440f));
}
__device__ __forceinline__ float sigmoidf(float x) {
    return 1.0f / (1.0f + expf(-x));
}

// fp32 -> f16 conversion (weights / staging), 4 elements per thread
__global__ __launch_bounds__(256) void cvt_f16_kernel(const float* __restrict__ in,
                                                      f16* __restrict__ out, int n)
{
    const int i4 = (blockIdx.x * 256 + threadIdx.x) * 4;
    if (i4 < n) {
        float4 v = *(const float4*)(in + i4);
        out[i4 + 0] = (f16)v.x; out[i4 + 1] = (f16)v.y;
        out[i4 + 2] = (f16)v.z; out[i4 + 3] = (f16)v.w;
    }
}

// ---------------------------------------------------------------------------
// WMMA GEMM:  C[M,N] = act(A[M,K] @ W[N,K]^T + bias), f16 operands, f32 accum.
// Wave computes a 16x64 strip (A reused across 4 WMMAs per k-step); block of
// 4 waves covers 64x64; grid (N/64, M/64). A loads: 2 x b128; B: 1 x 32B.
// Optional f32 and/or f16 outputs.
// ---------------------------------------------------------------------------
__global__ __launch_bounds__(128) void gemm_wmma(
    const f16* __restrict__ A, int lda,
    const f16* __restrict__ W, int ldw,
    const float* __restrict__ bias,
    float* __restrict__ C, f16* __restrict__ Ch, int ldc,
    int K, int act)
{
    const int lane = threadIdx.x & 31;
    const int wave = threadIdx.x >> 5;
    const int half = lane >> 4;
    const int l16  = lane & 15;
    const int m0 = blockIdx.y * 64 + wave * 16;
    const int n0 = blockIdx.x * 64;

    const f16* arow = A + (size_t)(m0 + l16) * lda;
    const f16* wr[4];
#pragma unroll
    for (int nt = 0; nt < 4; ++nt)
        wr[nt] = W + (size_t)(n0 + nt * 16 + l16) * ldw;

    v8f acc[4] = {};
    for (int k0 = 0; k0 < K; k0 += 32) {
        v16h a;
        v8h lo = *(const v8h*)(arow + k0 + half * 16);
        v8h hi = *(const v8h*)(arow + k0 + half * 16 + 8);
#pragma unroll
        for (int t = 0; t < 8; ++t) { a[t] = lo[t]; a[8 + t] = hi[t]; }
#pragma unroll
        for (int nt = 0; nt < 4; ++nt) {
            v16h b = *(const v16h*)(wr[nt] + k0 + half * 16);
            acc[nt] = __builtin_amdgcn_wmma_f32_16x16x32_f16(false, a, false, b,
                                                             (short)0, acc[nt], false, false);
        }
    }
#pragma unroll
    for (int nt = 0; nt < 4; ++nt) {
        const int n = n0 + nt * 16 + l16;
        const float bv = bias ? bias[n] : 0.0f;
#pragma unroll
        for (int r = 0; r < 8; ++r) {
            const int m = m0 + r + 8 * half;       // C/D layout rows r, r+8
            float v = acc[nt][r] + bv;
            if (act == 1) v = gelu_exact(v);
            if (C)  C[(size_t)m * ldc + n]  = v;
            if (Ch) Ch[(size_t)m * ldc + n] = (f16)v;
        }
    }
}

// ---------------------------------------------------------------------------
// Flash-style attention, one wave per (b, h, 16-query tile), d_head = 32.
// f16 q/k/v buffers; scores + PV via v_wmma_f32_16x16x32_f16; f16 output.
// ---------------------------------------------------------------------------
__global__ __launch_bounds__(32) void attn_wmma(
    const f16* __restrict__ qb, long long q_ts, long long q_bs,
    const f16* __restrict__ kb, long long k_ts, long long k_bs,
    const f16* __restrict__ vb, long long v_ts, long long v_bs,
    f16* __restrict__ ob, long long o_ts, long long o_bs,
    int Skv, float scale)
{
    const int lane = threadIdx.x;
    const int half = lane >> 4;
    const int l16  = lane & 15;
    const int q0 = blockIdx.x * 16;
    const int h  = blockIdx.y;
    const int b  = blockIdx.z;

    const f16* qp = qb + (size_t)b * q_bs + (size_t)h * DH;
    const f16* kp = kb + (size_t)b * k_bs + (size_t)h * DH;
    const f16* vp = vb + (size_t)b * v_bs + (size_t)h * DH;

    __shared__ _Float16 P[16][34];   // probabilities staged C-layout -> A-layout

    // Q tile, A-operand layout (M = query, K = head-dim 0..31): two b128 loads
    v16h aq;
    {
        const f16* qrow = qp + (size_t)(q0 + l16) * q_ts;
        v8h lo = *(const v8h*)(qrow + half * 16);
        v8h hi = *(const v8h*)(qrow + half * 16 + 8);
#pragma unroll
        for (int t = 0; t < 8; ++t) { aq[t] = lo[t]; aq[8 + t] = hi[t]; }
    }

    v8f o0 = {}, o1 = {};
    float mrun[8], lrun[8];
#pragma unroll
    for (int r = 0; r < 8; ++r) { mrun[r] = -1e30f; lrun[r] = 0.0f; }

    for (int key0 = 0; key0 < Skv; key0 += 32) {
        // K tiles, B-operand layout: one contiguous 32B vector per lane
        v16h bk0 = *(const v16h*)(kp + (size_t)(key0 + l16)      * k_ts + half * 16);
        v16h bk1 = *(const v16h*)(kp + (size_t)(key0 + 16 + l16) * k_ts + half * 16);
        v8f cz = {};
        v8f s0 = __builtin_amdgcn_wmma_f32_16x16x32_f16(false, aq, false, bk0,
                                                        (short)0, cz, false, false);
        v8f s1 = __builtin_amdgcn_wmma_f32_16x16x32_f16(false, aq, false, bk1,
                                                        (short)0, cz, false, false);
        // Online softmax over the 32 new keys; row m = r + 8*half lives across
        // the 16 lanes of one half -> shfl_xor masks 1,2,4,8 stay in-half.
#pragma unroll
        for (int r = 0; r < 8; ++r) {
            float a0 = s0[r] * scale, a1 = s1[r] * scale;
            float mx = fmaxf(a0, a1);
#pragma unroll
            for (int off = 1; off < 16; off <<= 1) mx = fmaxf(mx, __shfl_xor(mx, off, 32));
            const float mt = fmaxf(mrun[r], mx);
            const float alpha = expf(mrun[r] - mt);
            const float p0 = expf(a0 - mt), p1 = expf(a1 - mt);
            float rs = p0 + p1;
#pragma unroll
            for (int off = 1; off < 16; off <<= 1) rs += __shfl_xor(rs, off, 32);
            lrun[r] = lrun[r] * alpha + rs;
            mrun[r] = mt;
            o0[r] *= alpha;  o1[r] *= alpha;
            const int m = r + 8 * half;
            P[m][l16]      = (f16)p0;
            P[m][16 + l16] = (f16)p1;
        }
        __syncthreads();
        // P as A-operand (M = query, K = 32 keys of this block)
        v16h ap;
#pragma unroll
        for (int r = 0; r < 8; ++r) {
            const int kkp = aidx(r, half);
            ap[2*r]   = P[l16][kkp];
            ap[2*r+1] = P[l16][kkp + 1];
        }
        // V tiles, B-operand layout (K = key-in-block, N = out channel)
        v16h bv0, bv1;
#pragma unroll
        for (int r = 0; r < 8; ++r) {
            const int kkp = bidx(r, half);
            bv0[2*r]   = vp[(size_t)(key0 + kkp)     * v_ts + l16];
            bv0[2*r+1] = vp[(size_t)(key0 + kkp + 1) * v_ts + l16];
            bv1[2*r]   = vp[(size_t)(key0 + kkp)     * v_ts + 16 + l16];
            bv1[2*r+1] = vp[(size_t)(key0 + kkp + 1) * v_ts + 16 + l16];
        }
        o0 = __builtin_amdgcn_wmma_f32_16x16x32_f16(false, ap, false, bv0,
                                                    (short)0, o0, false, false);
        o1 = __builtin_amdgcn_wmma_f32_16x16x32_f16(false, ap, false, bv1,
                                                    (short)0, o1, false, false);
        __syncthreads();
    }

    f16* op = ob + (size_t)b * o_bs + (size_t)h * DH;
#pragma unroll
    for (int r = 0; r < 8; ++r) {
        const int m = q0 + r + 8 * half;
        const float inv = 1.0f / lrun[r];
        op[(size_t)m * o_ts + l16]      = (f16)(o0[r] * inv);
        op[(size_t)m * o_ts + 16 + l16] = (f16)(o1[r] * inv);
    }
}

// ---------------------------------------------------------------------------
// LayerNorm (optional residual, optional trailing GELU, optional f16 copy).
// ---------------------------------------------------------------------------
__global__ __launch_bounds__(256) void ln_kernel(
    const float* __restrict__ a, const float* __restrict__ res,
    const float* __restrict__ g, const float* __restrict__ bb,
    float* __restrict__ out, f16* __restrict__ outh, int post_gelu)
{
    const int t = blockIdx.x, d = threadIdx.x;
    __shared__ float red[256];
    float v = a[(size_t)t * DD + d];
    if (res) v += res[(size_t)t * DD + d];
    red[d] = v; __syncthreads();
    for (int s = 128; s; s >>= 1) { if (d < s) red[d] += red[d + s]; __syncthreads(); }
    const float mean = red[0] * (1.0f / DD); __syncthreads();
    const float c = v - mean;
    red[d] = c * c; __syncthreads();
    for (int s = 128; s; s >>= 1) { if (d < s) red[d] += red[d + s]; __syncthreads(); }
    const float var = red[0] * (1.0f / DD);
    float y = c * rsqrtf(var + 1e-5f) * g[d] + bb[d];
    if (post_gelu) y = gelu_exact(y);
    if (out)  out[(size_t)t * DD + d]  = y;
    if (outh) outh[(size_t)t * DD + d] = (f16)y;
}

// combine = gate*flow + (1-gate)*attn + bias, then LayerNorm(norm2)
__global__ __launch_bounds__(256) void combine_ln_kernel(
    const float* __restrict__ flow, const float* __restrict__ attn,
    const float* __restrict__ bias, const float* __restrict__ gatep,
    const float* __restrict__ g, const float* __restrict__ bb,
    float* __restrict__ out, f16* __restrict__ outh)
{
    const int t = blockIdx.x, d = threadIdx.x;
    __shared__ float red[256];
    const float gate = gatep[0];
    float v = gate * flow[(size_t)t * DD + d]
            + (1.0f - gate) * attn[(size_t)t * DD + d] + bias[d];
    red[d] = v; __syncthreads();
    for (int s = 128; s; s >>= 1) { if (d < s) red[d] += red[d + s]; __syncthreads(); }
    const float mean = red[0] * (1.0f / DD); __syncthreads();
    const float c = v - mean;
    red[d] = c * c; __syncthreads();
    for (int s = 128; s; s >>= 1) { if (d < s) red[d] += red[d + s]; __syncthreads(); }
    const float var = red[0] * (1.0f / DD);
    const float y = c * rsqrtf(var + 1e-5f) * g[d] + bb[d];
    out[(size_t)t * DD + d]  = y;
    outh[(size_t)t * DD + d] = (f16)y;
}

// ctx[b,:] = mean over sequence of x_norm
__global__ __launch_bounds__(256) void ctx_kernel(const float* __restrict__ xn,
                                                  float* __restrict__ ctx)
{
    const int b = blockIdx.x, d = threadIdx.x;
    float s = 0.0f;
    for (int t = 0; t < SS; ++t) s += xn[((size_t)b * SS + t) * DD + d];
    ctx[b * DD + d] = s * (1.0f / SS);
}

// window-attention scalar -> global adaptive-k (exact reference arithmetic)
__global__ __launch_bounds__(64) void win_kernel(
    const float* __restrict__ ctx,
    const float* __restrict__ w1, const float* __restrict__ b1,
    const float* __restrict__ w2, const float* __restrict__ b2,
    int* __restrict__ kkbuf)
{
    const int tid = threadIdx.x;
    __shared__ float red[64];
    __shared__ float winsum;
    if (tid == 0) winsum = 0.0f;
    __syncthreads();
    for (int b = 0; b < BB; ++b) {
        float s = b1[tid];
        const float* c = ctx + b * DD;
        for (int k = 0; k < DD; ++k) s += c[k] * w1[tid * DD + k];
        red[tid] = gelu_exact(s) * w2[tid];
        __syncthreads();
        for (int st = 32; st; st >>= 1) { if (tid < st) red[tid] += red[tid + st]; __syncthreads(); }
        if (tid == 0) {
            const float win = sigmoidf(red[0] + b2[0]) * (4096.0f - 256.0f) + 256.0f;
            winsum += win;
        }
        __syncthreads();
    }
    if (tid == 0) {
        const float spars = 0.1f * ((winsum * 0.5f) / 4096.0f);
        int kk = (int)floorf(65536.0f * spars);
        if (kk < 1) kk = 1;
        kkbuf[0] = kk;
    }
}

// router: pattern weights (softmax over 16) + flow intensity, one token/block
__global__ __launch_bounds__(64) void router_kernel(
    const float* __restrict__ xn, const float* __restrict__ ctx,
    const float* __restrict__ cs_w1, const float* __restrict__ cs_b1,
    const float* __restrict__ cs_w2, const float* __restrict__ cs_b2,
    const float* __restrict__ fi_w1, const float* __restrict__ fi_b1,
    const float* __restrict__ fi_w2, const float* __restrict__ fi_b2,
    float* __restrict__ pw, float* __restrict__ inten)
{
    const int token = blockIdx.x;
    const int b = token >> 9;    // token / S
    const int tid = threadIdx.x;
    __shared__ float comb[2 * DD];
    __shared__ float h1[32];
    __shared__ float lg[16];
    __shared__ float red[64];

    for (int i = tid; i < DD; i += 64) {
        comb[i]      = xn[(size_t)token * DD + i];
        comb[DD + i] = ctx[b * DD + i];
    }
    __syncthreads();
    if (tid < 32) {
        float s = cs_b1[tid];
        for (int k = 0; k < 2 * DD; ++k) s += comb[k] * cs_w1[tid * 2 * DD + k];
        h1[tid] = gelu_exact(s);
    }
    __syncthreads();
    if (tid < 16) {
        float s = cs_b2[tid];
        for (int j = 0; j < 32; ++j) s += h1[j] * cs_w2[tid * 32 + j];
        lg[tid] = s;
    }
    __syncthreads();
    if (tid == 0) {
        float mx = lg[0];
        for (int p = 1; p < PP; ++p) mx = fmaxf(mx, lg[p]);
        float e[PP]; float sum = 0.0f;
        for (int p = 0; p < PP; ++p) { e[p] = expf(lg[p] - mx); sum += e[p]; }
        const float inv = 1.0f / sum;
        for (int p = 0; p < PP; ++p) pw[(size_t)token * PP + p] = e[p] * inv;
    }
    // intensity MLP (64 hidden)
    {
        float s = fi_b1[tid];
        for (int k = 0; k < 2 * DD; ++k) s += comb[k] * fi_w1[tid * 2 * DD + k];
        red[tid] = gelu_exact(s) * fi_w2[tid];
    }
    __syncthreads();
    for (int st = 32; st; st >>= 1) { if (tid < st) red[tid] += red[tid + st]; __syncthreads(); }
    if (tid == 0) inten[token] = sigmoidf(red[0] + fi_b2[0]);
}

// ---------------------------------------------------------------------------
// Flow kernel: one workgroup = one token. The full 256KB flow matrix lives in
// LDS (320KB/WGP); exact k-th-largest via 4-round radix select on float bits
// (positive floats order as uints) with ds atomics; masked matvec from LDS.
// fm never touches HBM; patterns stream from L2 as float4 (b128) loads.
// ---------------------------------------------------------------------------
#define FM_PITCH 260
__global__ __launch_bounds__(512) void flow_kernel(
    const float* __restrict__ xn, const float* __restrict__ patterns,
    const float* __restrict__ pw, const float* __restrict__ inten,
    const int* __restrict__ kkbuf, float* __restrict__ flow)
{
    extern __shared__ float smem[];
    float*    fm   = smem;                       // 256 * 260
    float*    xrow = fm + DD * FM_PITCH;         // 256
    float*    psum = xrow + DD;                  // 512
    unsigned* hist = (unsigned*)(psum + 512);    // 256
    int*      bc   = (int*)(hist + 256);         // 2

    const int token = blockIdx.x;
    const int tid = threadIdx.x;

    const float iv = inten[token];
    float pwl[PP];
#pragma unroll
    for (int p = 0; p < PP; ++p) pwl[p] = pw[(size_t)token * PP + p] * iv;

    if (tid < DD) xrow[tid] = xn[(size_t)token * DD + tid];

    // build fm = intensity * sum_p pw_p * patterns_p  (65536 elems, float4)
    for (int e = tid * 4; e < DD * DD; e += 512 * 4) {
        const int i = e >> 8, j = e & 255;
        float vx = 0.0f, vy = 0.0f, vz = 0.0f, vw = 0.0f;
        __builtin_prefetch(patterns + e + 8192, 0, 1);
#pragma unroll
        for (int p = 0; p < PP; ++p) {
            float4 pv = *(const float4*)(patterns + (size_t)p * DD * DD + e);
            vx += pwl[p] * pv.x; vy += pwl[p] * pv.y;
            vz += pwl[p] * pv.z; vw += pwl[p] * pv.w;
        }
        float4 o; o.x = vx; o.y = vy; o.z = vz; o.w = vw;
        *(float4*)(&fm[i * FM_PITCH + j]) = o;
    }
    __syncthreads();

    // exact radix select of the kk-th largest |fm| (bit pattern of the value)
    const int kk = kkbuf[0];
    unsigned prefix = 0u;
    int remaining = kk;
    for (int round = 3; round >= 0; --round) {
        if (tid < 256) hist[tid] = 0u;
        __syncthreads();
        const int shift = round * 8;
        const unsigned hm = (round == 3) ? 0u : (0xFFFFFFFFu << (shift + 8));
        for (int e = tid * 4; e < DD * DD; e += 512 * 4) {
            const int i = e >> 8, j = e & 255;
            float4 v = *(const float4*)(&fm[i * FM_PITCH + j]);
            const unsigned u0 = __float_as_uint(fabsf(v.x));
            const unsigned u1 = __float_as_uint(fabsf(v.y));
            const unsigned u2 = __float_as_uint(fabsf(v.z));
            const unsigned u3 = __float_as_uint(fabsf(v.w));
            if ((u0 & hm) == prefix) atomicAdd(&hist[(u0 >> shift) & 255], 1u);
            if ((u1 & hm) == prefix) atomicAdd(&hist[(u1 >> shift) & 255], 1u);
            if ((u2 & hm) == prefix) atomicAdd(&hist[(u2 >> shift) & 255], 1u);
            if ((u3 & hm) == prefix) atomicAdd(&hist[(u3 >> shift) & 255], 1u);
        }
        __syncthreads();
        if (tid == 0) {
            int cum = 0, bsel = 0;
            for (int bkt = 255; bkt >= 0; --bkt) {
                const int c = (int)hist[bkt];
                if (cum + c >= remaining) { bsel = bkt; break; }
                cum += c;
            }
            bc[0] = bsel; bc[1] = remaining - cum;
        }
        __syncthreads();
        prefix |= ((unsigned)bc[0]) << shift;
        remaining = bc[1];
        __syncthreads();
    }
    const float thr = __uint_as_float(prefix);

    // masked matvec: flow[i] = sum_j (|fm_ij| >= thr) fm_ij * x_j   (float4)
    {
        const int i = tid >> 1;
        const int j0 = (tid & 1) * 128;
        float acc = 0.0f;
        for (int jj = 0; jj < 128; jj += 4) {
            float4 v  = *(const float4*)(&fm[i * FM_PITCH + j0 + jj]);
            float4 xv = *(const float4*)(&xrow[j0 + jj]);
            if (fabsf(v.x) >= thr) acc += v.x * xv.x;
            if (fabsf(v.y) >= thr) acc += v.y * xv.y;
            if (fabsf(v.z) >= thr) acc += v.z * xv.z;
            if (fabsf(v.w) >= thr) acc += v.w * xv.w;
        }
        psum[tid] = acc;
    }
    __syncthreads();
    if (tid < DD) flow[(size_t)token * DD + tid] = psum[2 * tid] + psum[2 * tid + 1];
}

// ---------------------------------------------------------------------------
extern "C" void kernel_launch(void* const* d_in, const int* in_sizes, int n_in,
                              void* d_out, int out_size, void* d_ws, size_t ws_size,
                              hipStream_t stream)
{
    const float* x        = (const float*)d_in[0];
    const float* n1g      = (const float*)d_in[1];
    const float* n1b      = (const float*)d_in[2];
    const float* n2g      = (const float*)d_in[3];
    const float* n2b      = (const float*)d_in[4];
    const float* n3g      = (const float*)d_in[5];
    const float* n3b      = (const float*)d_in[6];
    const float* fng      = (const float*)d_in[7];
    const float* fnb      = (const float*)d_in[8];
    const float* sa_in_w  = (const float*)d_in[9];
    const float* sa_in_b  = (const float*)d_in[10];
    const float* sa_out_w = (const float*)d_in[11];
    const float* sa_out_b = (const float*)d_in[12];
    const float* ca_in_w  = (const float*)d_in[13];
    const float* ca_in_b  = (const float*)d_in[14];
    const float* ca_out_w = (const float*)d_in[15];
    const float* ca_out_b = (const float*)d_in[16];
    const float* memory   = (const float*)d_in[17];
    const float* ffn_w1   = (const float*)d_in[18];
    const float* ffn_b1   = (const float*)d_in[19];
    const float* ffn_w2   = (const float*)d_in[20];
    const float* ffn_b2   = (const float*)d_in[21];
    const float* bias     = (const float*)d_in[22];
    const float* gate     = (const float*)d_in[23];
    const float* patterns = (const float*)d_in[24];
    const float* cs_w1    = (const float*)d_in[25];
    const float* cs_b1    = (const float*)d_in[26];
    const float* cs_w2    = (const float*)d_in[27];
    const float* cs_b2    = (const float*)d_in[28];
    const float* fi_w1    = (const float*)d_in[29];
    const float* fi_b1    = (const float*)d_in[30];
    const float* fi_w2    = (const float*)d_in[31];
    const float* fi_b2    = (const float*)d_in[32];
    const float* wa_w1    = (const float*)d_in[33];
    const float* wa_b1    = (const float*)d_in[34];
    const float* wa_w2    = (const float*)d_in[35];
    const float* wa_b2    = (const float*)d_in[36];
    float* out = (float*)d_out;

    // workspace carve-up (256B aligned chunks)
    float* w = (float*)d_ws;
    size_t off = 0;
    auto take  = [&](size_t n) { float* p = w + off; off += ((n + 63) / 64) * 64; return p; };
    auto takeh = [&](size_t n) { return (f16*)take((n + 1) / 2); };
    // fp32 buffers
    float* x_norm    = take((size_t)NTOK * DD);
    float* ctx       = take(BB * DD);
    float* pwb       = take((size_t)NTOK * PP);
    float* intenb    = take(NTOK);
    int*   kkbuf     = (int*)take(64);
    float* attn_out  = take((size_t)NTOK * DD);
    float* flowb     = take((size_t)NTOK * DD);
    float* comb1     = take((size_t)NTOK * DD);
    float* cross_out = take((size_t)NTOK * DD);
    float* comb2     = take((size_t)NTOK * DD);
    float* ffno      = take((size_t)NTOK * DD);
    // f16 staging buffers (activations + weights)
    f16* x_norm_h    = takeh((size_t)NTOK * DD);
    f16* comb1_h     = takeh((size_t)NTOK * DD);
    f16* comb2_h     = takeh((size_t)NTOK * DD);
    f16* qkv_h       = takeh((size_t)NTOK * 3 * DD);
    f16* attn_cat_h  = takeh((size_t)NTOK * DD);
    f16* kvmem_h     = takeh((size_t)MEMN * 2 * DD);
    f16* qcross_h    = takeh((size_t)NTOK * DD);
    f16* cross_cat_h = takeh((size_t)NTOK * DD);
    f16* ffnh_h      = takeh((size_t)NTOK * 4 * DD);
    f16* sa_in_w_h   = takeh(3 * DD * DD);
    f16* sa_out_w_h  = takeh(DD * DD);
    f16* ca_in_w_h   = takeh(3 * DD * DD);
    f16* ca_out_w_h  = takeh(DD * DD);
    f16* ffn_w1_h    = takeh(4 * DD * DD);
    f16* ffn_w2_h    = takeh(4 * DD * DD);
    f16* mem_h       = takeh(MEMN * DD);
    (void)ws_size; (void)n_in; (void)in_sizes; (void)out_size;

    const float scale = 1.0f / sqrtf((float)DH);

    // 0) weight conversions to f16 (L2-resident, one pass each)
    cvt_f16_kernel<<<3 * DD * DD / 1024, 256, 0, stream>>>(sa_in_w,  sa_in_w_h,  3 * DD * DD);
    cvt_f16_kernel<<<DD * DD / 1024,     256, 0, stream>>>(sa_out_w, sa_out_w_h, DD * DD);
    cvt_f16_kernel<<<3 * DD * DD / 1024, 256, 0, stream>>>(ca_in_w,  ca_in_w_h,  3 * DD * DD);
    cvt_f16_kernel<<<DD * DD / 1024,     256, 0, stream>>>(ca_out_w, ca_out_w_h, DD * DD);
    cvt_f16_kernel<<<4 * DD * DD / 1024, 256, 0, stream>>>(ffn_w1,   ffn_w1_h,   4 * DD * DD);
    cvt_f16_kernel<<<4 * DD * DD / 1024, 256, 0, stream>>>(ffn_w2,   ffn_w2_h,   4 * DD * DD);
    cvt_f16_kernel<<<MEMN * DD / 1024,   256, 0, stream>>>(memory,   mem_h,      MEMN * DD);

    // 1) x_norm = LN1(x)  (+ f16 copy for WMMA consumers)
    ln_kernel<<<NTOK, 256, 0, stream>>>(x, nullptr, n1g, n1b, x_norm, x_norm_h, 0);
    // 2) ctx = mean_s(x_norm)
    ctx_kernel<<<BB, 256, 0, stream>>>(x_norm, ctx);
    // 3) adaptive-k scalar
    win_kernel<<<1, 64, 0, stream>>>(ctx, wa_w1, wa_b1, wa_w2, wa_b2, kkbuf);
    // 4) router: pattern weights + intensity
    router_kernel<<<NTOK, 64, 0, stream>>>(x_norm, ctx, cs_w1, cs_b1, cs_w2, cs_b2,
                                           fi_w1, fi_b1, fi_w2, fi_b2, pwb, intenb);
    // 5) self-attn QKV projection  [1024,256] x [256,768] -> f16
    gemm_wmma<<<dim3(768 / 64, NTOK / 64), 128, 0, stream>>>(
        x_norm_h, DD, sa_in_w_h, DD, sa_in_b, nullptr, qkv_h, 3 * DD, DD, 0);
    // 6) self attention core
    attn_wmma<<<dim3(SS / 16, HH, BB), 32, 0, stream>>>(
        qkv_h,          3 * DD, (long long)SS * 3 * DD,
        qkv_h + DD,     3 * DD, (long long)SS * 3 * DD,
        qkv_h + 2 * DD, 3 * DD, (long long)SS * 3 * DD,
        attn_cat_h,     DD,     (long long)SS * DD,
        SS, scale);
    // 7) self-attn output projection -> f32
    gemm_wmma<<<dim3(DD / 64, NTOK / 64), 128, 0, stream>>>(
        attn_cat_h, DD, sa_out_w_h, DD, sa_out_b, attn_out, nullptr, DD, DD, 0);
    // 8) flow path: full fm row in LDS, radix-select threshold, masked matvec
    {
        const size_t shmem = (size_t)(DD * FM_PITCH + DD + 512) * 4 + 256 * 4 + 16;
        flow_kernel<<<NTOK, 512, shmem, stream>>>(x_norm, patterns, pwb, intenb,
                                                  kkbuf, flowb);
    }
    // 9) combine + LN2
    combine_ln_kernel<<<NTOK, 256, 0, stream>>>(flowb, attn_out, bias, gate,
                                                n2g, n2b, comb1, comb1_h);
    // 10) memory K/V projections (batch-shared) and cross-Q -> f16
    gemm_wmma<<<dim3(DD / 64, MEMN / 64), 128, 0, stream>>>(
        mem_h, DD, ca_in_w_h + DD * DD, DD, ca_in_b + DD,
        nullptr, kvmem_h, 2 * DD, DD, 0);
    gemm_wmma<<<dim3(DD / 64, MEMN / 64), 128, 0, stream>>>(
        mem_h, DD, ca_in_w_h + 2 * DD * DD, DD, ca_in_b + 2 * DD,
        nullptr, kvmem_h + DD, 2 * DD, DD, 0);
    gemm_wmma<<<dim3(DD / 64, NTOK / 64), 128, 0, stream>>>(
        comb1_h, DD, ca_in_w_h, DD, ca_in_b, nullptr, qcross_h, DD, DD, 0);
    // 11) cross attention (kv shared across batch: batch stride 0)
    attn_wmma<<<dim3(SS / 16, HH, BB), 32, 0, stream>>>(
        qcross_h,      DD,     (long long)SS * DD,
        kvmem_h,       2 * DD, 0LL,
        kvmem_h + DD,  2 * DD, 0LL,
        cross_cat_h,   DD,     (long long)SS * DD,
        MEMN, scale);
    // 12) cross-attn output projection -> f32
    gemm_wmma<<<dim3(DD / 64, NTOK / 64), 128, 0, stream>>>(
        cross_cat_h, DD, ca_out_w_h, DD, ca_out_b, cross_out, nullptr, DD, DD, 0);
    // 13) LN3(comb1 + cross_out)  (+ f16 copy)
    ln_kernel<<<NTOK, 256, 0, stream>>>(comb1, cross_out, n3g, n3b, comb2, comb2_h, 0);
    // 14) FFN: hidden = gelu(comb2 @ w1^T + b1) -> f16  [1024,256]x[256,1024]
    gemm_wmma<<<dim3(4 * DD / 64, NTOK / 64), 128, 0, stream>>>(
        comb2_h, DD, ffn_w1_h, DD, ffn_b1, nullptr, ffnh_h, 4 * DD, DD, 1);
    // 15) ffn_out = hidden @ w2^T + b2 -> f32   [1024,1024]x[1024,256]
    gemm_wmma<<<dim3(DD / 64, NTOK / 64), 128, 0, stream>>>(
        ffnh_h, 4 * DD, ffn_w2_h, 4 * DD, ffn_b2, ffno, nullptr, DD, 4 * DD, 0);
    // 16) out = gelu(LNf(comb2 + ffn_out))
    ln_kernel<<<NTOK, 256, 0, stream>>>(comb2, ffno, fng, fnb, out, nullptr, 1);
}
